// MultiheadAttention_16767552324020
// MI455X (gfx1250) — matrix operand order repaired and checked by reading, not compiled, not run on hardware
//
#include <hip/hip_runtime.h>
#include <hip/hip_bf16.h>
#include <math.h>

// ---------------------------------------------------------------------------
// Problem constants (from reference): B=2, S=2048, C=1536, H=8, DK=64, DV=192
// ---------------------------------------------------------------------------
#define BATCH 2
#define SEQ   2048
#define CDIM  1536
#define HEADS 8
#define DK    64
#define DV    192
#define NPOS  4095          // 2S-1
#define NPOSP 4096          // padded rows for GEMM tiling
#define NB    32            // basis per class (F_REL/6)

typedef __bf16 bf16;
typedef __attribute__((ext_vector_type(16))) __bf16 v16bf;
typedef __attribute__((ext_vector_type(8)))  __bf16 v8bf;
typedef __attribute__((ext_vector_type(8)))  float  v8f;

// ------------------------------- helpers -----------------------------------
// A-fragment (16x32 bf16, M in lanes 0..15): two 16B loads per lane.
// rowp must already point at this lane's row start + k0 offset.
__device__ __forceinline__ v16bf frag_a_row(const bf16* rowp, int hi) {
  union { v16bf v; v8bf h[2]; } u;
  u.h[0] = *(const v8bf*)(rowp + 8 * hi);
  u.h[1] = *(const v8bf*)(rowp + 16 + 8 * hi);
  return u.v;
}

// B-fragment (32x16 bf16) read from an [N][K]-major (transposed) buffer:
// lane lc -> column n0+lc, elements = K k0+8*hi..+7, then k0+16+8*hi..+7.
template <int STRIDE>
__device__ __forceinline__ v16bf frag_b_nk(const bf16* basep, int n0, int k0, int lane) {
  const int lc = lane & 15, hi = lane >> 4;
  const bf16* p = basep + (size_t)(n0 + lc) * STRIDE + k0 + 8 * hi;
  union { v16bf v; v8bf h[2]; } u;
  u.h[0] = *(const v8bf*)(p);
  u.h[1] = *(const v8bf*)(p + 16);
  return u.v;
}

__device__ __forceinline__ v8f wmma_bf16(v16bf a, v16bf b, v8f c) {
  return __builtin_amdgcn_wmma_f32_16x16x32_bf16(
      /*neg_a=*/false, a, /*neg_b=*/false, b,
      /*c_mod=*/(short)0, c, /*reuse_a=*/false, /*reuse_b=*/false);
}

// CDNA5 async global->LDS copy, 16 bytes per lane, tracked by ASYNCcnt.
__device__ __forceinline__ void async_copy16(unsigned lds_off, const void* gptr) {
  unsigned long long ga = (unsigned long long)(size_t)gptr;
  asm volatile("global_load_async_to_lds_b128 %0, %1, off"
               :: "v"(lds_off), "v"(ga) : "memory");
}
__device__ __forceinline__ void async_wait0() {
  asm volatile("s_wait_asynccnt 0x0" ::: "memory");
}

// ---------------------- positional features --------------------------------
__device__ __forceinline__ float gamma_prob(float x, int j) {
  // stddev = S/(2n) = 32 ; mean = linspace(S/n=64, S=2048, 32)
  float mean = 64.0f + (float)j * (1984.0f / 31.0f);
  float conc = (mean * (1.0f / 32.0f)) * (mean * (1.0f / 32.0f));
  float rate = mean * (1.0f / 1024.0f);
  float lnorm = lgammaf(conc) - conc * logf(rate);
  float lu = (x > 0.0f) ? (conc - 1.0f) * logf(x) - rate * x : -1.0e30f;
  return __expf(lu - lnorm) + 1e-8f;
}

__global__ void gmax_kernel(float* __restrict__ gmax) {
  int j = threadIdx.x;                 // 32 threads, one basis each
  float mx = 0.0f;
  for (int x = 0; x < SEQ; ++x) mx = fmaxf(mx, gamma_prob((float)x, j));
  gmax[j] = mx;
}

__global__ void posfeat_kernel(const float* __restrict__ gmax, bf16* __restrict__ pe) {
  int idx = blockIdx.x * blockDim.x + threadIdx.x;   // NPOSP * 192 threads
  int p = idx / 192, f = idx % 192;
  float v = 0.0f;
  if (p < NPOS) {
    int dist = p - (SEQ - 1);
    float ap = fabsf((float)dist);
    int bidx = f % 96, cls = bidx / 32, j = bidx % 32;
    if (cls == 0) {                               // exponential decay
      float hl = exp2f(3.0f + (float)j * (8.0f / 31.0f));  // 2^linspace(3,11,32)
      v = exp2f(-ap / hl);
    } else if (cls == 1) {                        // central mask
      float cw = exp2f((float)(j + 1)) - 1.0f;
      v = (cw > ap) ? 1.0f : 0.0f;
    } else {                                      // gamma pdf, max-normalized
      v = gamma_prob(ap, j) / gmax[j];
    }
    if (f >= 96) {
      float sgn = (dist > 0) ? 1.0f : ((dist < 0) ? -1.0f : 0.0f);
      v *= sgn;
    }
  }
  pe[idx] = (bf16)v;
}

// --------------------------- generic WMMA GEMM ------------------------------
// C[M,N] = A[M,K] @ B[K,N]; A is f32 or bf16 global row-major (template flag),
// B is f32 global row-major, converted to bf16 while staging into LDS.
// Tile 128x128, 256 threads / 8 waves, each wave 16 rows x 128 cols.
// MODE 0: out0 = bf16(C)
// MODE 1: out0 = bf16(C*scale), out1 = bf16(C*scale + bias[col])
// MODE 2: outf = C + bias[col] (f32)
template <int MODE, bool ABF>
__global__ __launch_bounds__(256) void gemm_kernel(
    const void* __restrict__ Aptr, const float* __restrict__ Bw,
    int K, int lda, int ldb, float scale, const float* __restrict__ bias,
    bf16* __restrict__ out0, bf16* __restrict__ out1,
    float* __restrict__ outf, int ldo) {
  __shared__ bf16 Al[128][32];   // [M][K] tile
  __shared__ bf16 Bt[128][32];   // [N][K] (transposed) tile
  const int tid = threadIdx.x, wid = tid >> 5, lane = tid & 31;
  const int hi = lane >> 4, lc = lane & 15;
  const int m0 = blockIdx.y * 128, n0 = blockIdx.x * 128;

  v8f zero = {};
  v8f acc[8];
#pragma unroll
  for (int t = 0; t < 8; ++t) acc[t] = zero;

  for (int k0 = 0; k0 < K; k0 += 32) {
    __syncthreads();
    if constexpr (ABF) {
      const bf16* A = (const bf16*)Aptr;
#pragma unroll
      for (int i = 0; i < 2; ++i) {                    // 512 x 16B chunks
        int idx = i * 256 + tid;
        int r = idx >> 2, c8 = (idx & 3) * 8;
        *(uint4*)&Al[r][c8] = *(const uint4*)(A + (size_t)(m0 + r) * lda + k0 + c8);
      }
    } else {
      const float* A = (const float*)Aptr;
#pragma unroll
      for (int i = 0; i < 4; ++i) {                    // 1024 x float4
        int idx = i * 256 + tid;
        int r = idx >> 3, c4 = (idx & 7) * 4;
        float4 f = *(const float4*)(A + (size_t)(m0 + r) * lda + k0 + c4);
        union { bf16 b[4]; unsigned long long u; } w;
        w.b[0] = (bf16)f.x; w.b[1] = (bf16)f.y; w.b[2] = (bf16)f.z; w.b[3] = (bf16)f.w;
        *(unsigned long long*)&Al[r][c4] = w.u;
      }
    }
#pragma unroll
    for (int i = 0; i < 4; ++i) {                      // B: 32 rows x 128 cols
      int idx = i * 256 + tid;
      int kk = idx >> 5, n4 = (idx & 31) * 4;
      float4 f = *(const float4*)(Bw + (size_t)(k0 + kk) * ldb + n0 + n4);
      Bt[n4 + 0][kk] = (bf16)f.x;
      Bt[n4 + 1][kk] = (bf16)f.y;
      Bt[n4 + 2][kk] = (bf16)f.z;
      Bt[n4 + 3][kk] = (bf16)f.w;
    }
    if (k0 + 32 < K) {                                 // prefetch next K-tile
      __builtin_prefetch(Bw + (size_t)(k0 + 32 + (tid & 31)) * ldb + n0 + (tid >> 5) * 16, 0, 0);
      if constexpr (ABF)
        __builtin_prefetch((const bf16*)Aptr + (size_t)(m0 + (tid >> 1)) * lda + k0 + 32, 0, 0);
      else
        __builtin_prefetch((const float*)Aptr + (size_t)(m0 + (tid >> 1)) * lda + k0 + 32, 0, 0);
    }
    __syncthreads();
    v16bf a = frag_a_row(&Al[16 * wid + lc][0], hi);
#pragma unroll
    for (int tn = 0; tn < 8; ++tn)
      acc[tn] = wmma_bf16(a, frag_b_nk<32>(&Bt[0][0], 16 * tn, 0, lane), acc[tn]);
  }

#pragma unroll
  for (int tn = 0; tn < 8; ++tn)
#pragma unroll
    for (int e = 0; e < 8; ++e) {
      int r = e + 8 * hi;
      int row = m0 + 16 * wid + r, col = n0 + 16 * tn + lc;
      float v = acc[tn][e];
      size_t oi = (size_t)row * ldo + col;
      if constexpr (MODE == 0) {
        out0[oi] = (bf16)v;
      } else if constexpr (MODE == 1) {
        float s = v * scale;
        out0[oi] = (bf16)s;
        out1[oi] = (bf16)(s + bias[col]);
      } else {
        outf[oi] = v + bias[col];
      }
    }
}

// ------------------------- flash attention w/ rel shift ---------------------
// One workgroup (64 thr = 2 waves) handles (b, h, 32 q-rows); loops 64-key blocks.
// K and r_k tiles are staged with CDNA5 async global->LDS copies (ASYNCcnt),
// V is staged transposed (vector global reads, scalar LDS scatter).
__global__ __launch_bounds__(64) void attn_kernel(
    const bf16* __restrict__ qc, const bf16* __restrict__ qr,
    const bf16* __restrict__ kmat, const bf16* __restrict__ vmat,
    const bf16* __restrict__ rk, const int* __restrict__ mask,
    bf16* __restrict__ attn) {
  __shared__ bf16 Kb[64][64];        // [key][dk]
  __shared__ bf16 VbT[192][64];      // [vdim][key] (transposed for B-frags)
  __shared__ bf16 Rk[96][64];        // [m][dk], rows base..base+95
  __shared__ float Pls[2][16][96];   // per-wave rel-logit scratch (pre-shift)
  __shared__ bf16 Pp[2][16][64];     // per-wave probs for PV A-frags
  __shared__ int msk[64];

  const int tid = threadIdx.x;
  const int wid = tid >> 5, lane = tid & 31;
  const int hi = lane >> 4, lc = lane & 15;
  const int b = blockIdx.z, h = blockIdx.y, q0 = blockIdx.x * 32;

  // Per-lane async-copy geometry: 8 lanes cover one 128B row; 8 rows per pass.
  const int arow = wid * 4 + (lane >> 3);   // row within a pass (0..7)
  const int aoff = (lane & 7) * 16;         // byte offset within row
  const unsigned kb_lds = (unsigned)(size_t)&Kb[0][0];
  const unsigned rk_lds = (unsigned)(size_t)&Rk[0][0];

  // Q fragments (content + rel) live in registers for the whole kernel.
  v16bf qcf[2], qrf[2];
  {
    const size_t grow = (size_t)(b * SEQ + q0 + 16 * wid + lc);
    const bf16* pc = qc + grow * (HEADS * DK) + h * DK;
    const bf16* pr = qr + grow * (HEADS * DK) + h * DK;
#pragma unroll
    for (int ks = 0; ks < 2; ++ks) {
      qcf[ks] = frag_a_row(pc + 32 * ks, hi);
      qrf[ks] = frag_a_row(pr + 32 * ks, hi);
    }
  }

  float mrow[8], lrow[8];
  v8f zero = {};
  v8f Oacc[12];
#pragma unroll
  for (int e = 0; e < 8; ++e) { mrow[e] = -3.0e38f; lrow[e] = 0.0f; }
#pragma unroll
  for (int vn = 0; vn < 12; ++vn) Oacc[vn] = zero;

  for (int kbi = 0; kbi < SEQ / 64; ++kbi) {
    const int j0 = kbi * 64;
    const int base = (SEQ - 1) + j0 - q0 - 31;   // rel-row band start (>=0)
    __syncthreads();
    // ---- stage K tile [64][64] via async copies (8 rows per pass)
#pragma unroll
    for (int it = 0; it < 8; ++it) {
      int row = it * 8 + arow;
      async_copy16(kb_lds + row * 128 + aoff,
                   (const char*)(kmat + (size_t)(b * SEQ + j0 + row) * (HEADS * DK) + h * DK) + aoff);
    }
    // ---- stage r_k band [96][64] via async copies
#pragma unroll
    for (int it = 0; it < 12; ++it) {
      int row = it * 8 + arow;
      async_copy16(rk_lds + row * 128 + aoff,
                   (const char*)(rk + (size_t)(base + row) * (HEADS * DK) + h * DK) + aoff);
    }
    // ---- stage V tile transposed [192][64]: wide reads, scatter stores
#pragma unroll
    for (int it = 0; it < 24; ++it) {            // 12288 elems / (64 thr * 8)
      int idx = (it * 64 + tid) * 8;
      int key = idx / 192, vd = idx % 192;       // 8-chunks never cross a row
      union { uint4 u; bf16 bv[8]; } t;
      t.u = *(const uint4*)(vmat + (size_t)(b * SEQ + j0 + key) * (HEADS * DV) + h * DV + vd);
#pragma unroll
      for (int e = 0; e < 8; ++e) VbT[vd + e][key] = t.bv[e];
    }
    msk[tid] = mask[b * SEQ + j0 + tid];
    async_wait0();
    __syncthreads();

    // ---- content logits: 4 col-tiles per wave's 16 rows
    v8f Ct[4];
#pragma unroll
    for (int tn = 0; tn < 4; ++tn) {
      v8f a = zero;
#pragma unroll
      for (int ks = 0; ks < 2; ++ks)
        a = wmma_bf16(qcf[ks], frag_b_nk<64>(&Kb[0][0], 16 * tn, 32 * ks, lane), a);
      Ct[tn] = a;
    }
    // ---- rel logits (pre-shift): 16x96 per wave -> LDS
#pragma unroll
    for (int pn = 0; pn < 6; ++pn) {
      v8f a = zero;
#pragma unroll
      for (int ks = 0; ks < 2; ++ks)
        a = wmma_bf16(qrf[ks], frag_b_nk<64>(&Rk[0][0], 16 * pn, 32 * ks, lane), a);
#pragma unroll
      for (int e = 0; e < 8; ++e) Pls[wid][e + 8 * hi][16 * pn + lc] = a[e];
    }
    __syncthreads();

    // ---- combine content + shifted rel, mask, online softmax ---------------
    float bm[8];
#pragma unroll
    for (int e = 0; e < 8; ++e) bm[e] = -3.0e38f;
#pragma unroll
    for (int tn = 0; tn < 4; ++tn) {
      const int jl = 16 * tn + lc;
      const int mz = msk[jl];
#pragma unroll
      for (int e = 0; e < 8; ++e) {
        int r = e + 8 * hi;
        int m = jl + 31 - (16 * wid + r);         // relative-shift gather
        float lv = Ct[tn][e] + Pls[wid][r][m];
        lv = mz ? lv : -1e9f;
        bm[e] = fmaxf(bm[e], lv);
      }
    }
#pragma unroll
    for (int off = 8; off; off >>= 1)
#pragma unroll
      for (int e = 0; e < 8; ++e)
        bm[e] = fmaxf(bm[e], __shfl_xor(bm[e], off, 32));  // 16-lane row reduce

    float alpha[8], ssum[8];
#pragma unroll
    for (int e = 0; e < 8; ++e) {
      float mn = fmaxf(mrow[e], bm[e]);
      alpha[e] = __expf(mrow[e] - mn);
      mrow[e] = mn;
      ssum[e] = 0.0f;
    }
#pragma unroll
    for (int tn = 0; tn < 4; ++tn) {
      const int jl = 16 * tn + lc;
      const int mz = msk[jl];
#pragma unroll
      for (int e = 0; e < 8; ++e) {
        int r = e + 8 * hi;
        int m = jl + 31 - (16 * wid + r);
        float lv = Ct[tn][e] + Pls[wid][r][m];
        lv = mz ? lv : -1e9f;
        float p = __expf(lv - mrow[e]);
        ssum[e] += p;
        Pp[wid][r][jl] = (bf16)p;
      }
    }
#pragma unroll
    for (int off = 8; off; off >>= 1)
#pragma unroll
      for (int e = 0; e < 8; ++e) ssum[e] += __shfl_xor(ssum[e], off, 32);
#pragma unroll
    for (int e = 0; e < 8; ++e) lrow[e] = lrow[e] * alpha[e] + ssum[e];
#pragma unroll
    for (int vn = 0; vn < 12; ++vn)
#pragma unroll
      for (int e = 0; e < 8; ++e) Oacc[vn][e] = Oacc[vn][e] * alpha[e];
    __syncthreads();

    // ---- O += P @ V : 12 col-tiles of 16, K=64 keys (2 k-steps) ------------
#pragma unroll
    for (int vn = 0; vn < 12; ++vn) {
      v8f a = Oacc[vn];
#pragma unroll
      for (int ks = 0; ks < 2; ++ks) {
        v16bf pf = frag_a_row(&Pp[wid][lc][32 * ks], hi);
        a = wmma_bf16(pf, frag_b_nk<64>(&VbT[0][0], 16 * vn, 32 * ks, lane), a);
      }
      Oacc[vn] = a;
    }
  }

  // ---- epilogue: O / l -> bf16 attn[B*S][H*DV] -----------------------------
#pragma unroll
  for (int vn = 0; vn < 12; ++vn)
#pragma unroll
    for (int e = 0; e < 8; ++e) {
      int r = e + 8 * hi;
      float o = Oacc[vn][e] / lrow[e];
      attn[(size_t)(b * SEQ + q0 + 16 * wid + r) * (HEADS * DV) + h * DV + 16 * vn + lc] =
          (bf16)o;
    }
}

// ------------------------------- launcher -----------------------------------
extern "C" void kernel_launch(void* const* d_in, const int* in_sizes, int n_in,
                              void* d_out, int out_size, void* d_ws, size_t ws_size,
                              hipStream_t stream) {
  const float* x   = (const float*)d_in[0];
  const int*   msk = (const int*)d_in[1];
  const float* Wq  = (const float*)d_in[2];
  const float* Wk  = (const float*)d_in[3];
  const float* Wv  = (const float*)d_in[4];
  const float* Wr  = (const float*)d_in[5];
  const float* rwb = (const float*)d_in[6];   // [H*DK] = 512 bias vector
  const float* Wo  = (const float*)d_in[7];
  const float* bo  = (const float*)d_in[8];
  float* out = (float*)d_out;

  const int M = BATCH * SEQ;                  // 4096
  char* ws = (char*)d_ws;
  auto carve = [&](size_t bytes) {
    char* p = ws;
    ws += (bytes + 255) & ~(size_t)255;
    return p;
  };
  float* gmax  = (float*)carve(NB * sizeof(float));
  bf16* pe_b   = (bf16*)carve((size_t)NPOSP * 192 * 2);
  bf16* rk_b   = (bf16*)carve((size_t)NPOSP * (HEADS * DK) * 2);
  bf16* qc_b   = (bf16*)carve((size_t)M * (HEADS * DK) * 2);
  bf16* qr_b   = (bf16*)carve((size_t)M * (HEADS * DK) * 2);
  bf16* k_b    = (bf16*)carve((size_t)M * (HEADS * DK) * 2);
  bf16* v_b    = (bf16*)carve((size_t)M * (HEADS * DV) * 2);
  bf16* attn_b = (bf16*)carve((size_t)M * (HEADS * DV) * 2);
  (void)in_sizes; (void)n_in; (void)out_size; (void)ws_size;

  // 1) positional features
  gmax_kernel<<<1, 32, 0, stream>>>(gmax);
  posfeat_kernel<<<(NPOSP * 192) / 256, 256, 0, stream>>>(gmax, pe_b);

  // 2) r_k = pe @ Wr   [4096,192]x[192,512]
  gemm_kernel<0, true><<<dim3(512 / 128, NPOSP / 128), 256, 0, stream>>>(
      pe_b, Wr, 192, 192, 512, 1.0f, nullptr, rk_b, nullptr, nullptr, 512);

  // 3) projections   q (scaled + biased copy), k, v
  gemm_kernel<1, false><<<dim3(512 / 128, M / 128), 256, 0, stream>>>(
      x, Wq, CDIM, CDIM, 512, 0.125f, rwb, qc_b, qr_b, nullptr, 512);
  gemm_kernel<0, false><<<dim3(512 / 128, M / 128), 256, 0, stream>>>(
      x, Wk, CDIM, CDIM, 512, 1.0f, nullptr, k_b, nullptr, nullptr, 512);
  gemm_kernel<0, false><<<dim3(1536 / 128, M / 128), 256, 0, stream>>>(
      x, Wv, CDIM, CDIM, 1536, 1.0f, nullptr, v_b, nullptr, nullptr, 1536);

  // 4) flash attention with relative shift
  attn_kernel<<<dim3(SEQ / 32, HEADS, BATCH), 64, 0, stream>>>(
      qc_b, qr_b, k_b, v_b, rk_b, msk, attn_b);

  // 5) out = attn @ Wo + bo  -> f32
  gemm_kernel<2, true><<<dim3(1536 / 128, M / 128), 256, 0, stream>>>(
      attn_b, Wo, CDIM, 1536, 1536, 1.0f, bo, nullptr, nullptr, out, 1536);
}
